// MultiHeadAttentionLayer_63393717289370
// MI455X (gfx1250) — matrix operand order repaired and checked by reading, not compiled, not run on hardware
//
#include <hip/hip_runtime.h>

// ---------------------------------------------------------------------------
// MHA for B=2, S=2048, D_MODEL=1024, H=16, depth=64 on gfx1250 (CDNA5).
// bf16 WMMA (v_wmma_f32_16x16x32_bf16), f32 accumulation, flash attention,
// double-buffered GEMM with async global->LDS staging when available.
// ---------------------------------------------------------------------------

#define DM     1024
#define SEQ    2048
#define BATCH  2
#define HEADS  16
#define DEPTH  64
#define MROWS  (BATCH * SEQ)   // 4096

typedef unsigned short u16;
typedef __attribute__((ext_vector_type(4)))  int    v4i;
typedef __attribute__((ext_vector_type(8)))  __bf16 v8bf;
typedef __attribute__((ext_vector_type(16))) __bf16 v16bf;
typedef __attribute__((ext_vector_type(8)))  float  v8f;

#if defined(__AMDGCN__) && __has_builtin(__builtin_amdgcn_global_load_async_to_lds_b128)
#define ASYNC_LDS 1
#endif

__device__ __forceinline__ u16 f2bf(float f) {
    __bf16 h = (__bf16)f;
    return __builtin_bit_cast(unsigned short, h);
}

__device__ __forceinline__ v8bf ld8(const u16* p) {
    return *(const v8bf*)p;
}

__device__ __forceinline__ v16bf cat16(v8bf lo, v8bf hi) {
    return __builtin_shufflevector(lo, hi, 0,1,2,3,4,5,6,7,8,9,10,11,12,13,14,15);
}

__device__ __forceinline__ v8f wmma_bf16(v16bf a, v16bf b, v8f c) {
    return __builtin_amdgcn_wmma_f32_16x16x32_bf16(
        /*neg_a=*/false, a, /*neg_b=*/false, b,
        /*c_mod=*/(short)0, c, /*reuse_a=*/false, /*reuse_b=*/false);
}

#ifdef ASYNC_LDS
typedef __attribute__((address_space(1))) v4i* gptr4;   // global int4*
typedef __attribute__((address_space(3))) v4i* lptr4;   // LDS int4*

// async 16-byte global -> LDS copy (per-lane addresses), tracked by ASYNCcnt
__device__ __forceinline__ void async_cp16(const u16* g, u16* l) {
    v4i* gg = (v4i*)(uintptr_t)g;   // strip const, keep generic
    v4i* ll = (v4i*)l;
    __builtin_amdgcn_global_load_async_to_lds_b128(
        (gptr4)gg, (lptr4)ll, /*imm offset=*/0, /*cpol=*/0);
}
__device__ __forceinline__ void wait_async_lds() {
#if __has_builtin(__builtin_amdgcn_s_wait_asynccnt)
    __builtin_amdgcn_s_wait_asynccnt(0);
#else
    asm volatile("s_wait_asynccnt 0x0" ::: "memory");
#endif
}
#endif

// ---------------------------------------------------------------------------
// f32 -> bf16 conversion (vectorized by 4)
// ---------------------------------------------------------------------------
__global__ __launch_bounds__(256) void cvt_bf16(const float* __restrict__ s,
                                                u16* __restrict__ d, int n) {
    int i = (blockIdx.x * blockDim.x + threadIdx.x) * 4;
    if (i < n) {
        float4 f = *(const float4*)(s + i);
        ushort4 o;
        o.x = f2bf(f.x); o.y = f2bf(f.y); o.z = f2bf(f.z); o.w = f2bf(f.w);
        *(ushort4*)(d + i) = o;
    }
}

// ---------------------------------------------------------------------------
// GEMM core: C[4096x1024] = A[4096x1024](bf16) * W[1024x1024](bf16) + bias
// Workgroup tile 128(M) x 64(N), 8 waves, each wave 32x32 (2x2 WMMA frags).
// Double-buffered K-loop: async-stage tile k+1 while computing tile k,
// one barrier per iteration. W tile stored transposed [n][k] so B fragments
// are contiguous 32B ds_load_b128 pairs; padded strides avoid bank conflicts.
// ---------------------------------------------------------------------------
#define SA_STRIDE 40
__device__ __forceinline__ void gemm_core(const u16* __restrict__ A,
                                          const u16* __restrict__ W,
                                          const float* __restrict__ bias,
                                          float* __restrict__ outF,
                                          u16* __restrict__ outB) {
    __shared__ u16 sA[2][128 * SA_STRIDE];
    __shared__ u16 sB[2][64 * SA_STRIDE];

    const int tid  = threadIdx.x;
    const int lane = tid & 31;
    const int wave = tid >> 5;
    const int wm   = wave & 3;       // 0..3 -> M offset wm*32
    const int wn   = wave >> 2;      // 0..1 -> N offset wn*32
    const int lh   = lane >> 4;      // half-wave id (0/1)
    const int ln   = lane & 15;
    const int blockN = blockIdx.x * 64;
    const int blockM = blockIdx.y * 128;

    // staging coords
    const int arow = tid >> 1;           // 0..127
    const int aseg = (tid & 1) * 16;     // 0 / 16
    const int bk   = tid >> 3;           // 0..31
    const int bn   = (tid & 7) * 8;      // 0..56

    v8f acc[2][2] = {};

    // ---- prologue: stage k0 = 0 into buffer 0 ----
    {
        const u16* ap = A + (size_t)(blockM + arow) * DM + aseg;
        u16* la = &sA[0][arow * SA_STRIDE + aseg];
#ifdef ASYNC_LDS
        async_cp16(ap, la);
        async_cp16(ap + 8, la + 8);
#else
        *(v8bf*)la       = ld8(ap);
        *(v8bf*)(la + 8) = ld8(ap + 8);
#endif
        v8bf b0 = ld8(W + (size_t)bk * DM + blockN + bn);
        #pragma unroll
        for (int j = 0; j < 8; ++j)
            sB[0][(bn + j) * SA_STRIDE + bk] = __builtin_bit_cast(u16, b0[j]);
#ifdef ASYNC_LDS
        wait_async_lds();
#endif
        __syncthreads();
    }

    int cur = 0;
    for (int k0 = 0; k0 < DM; k0 += 32) {
        const int  nxt      = cur ^ 1;
        const bool has_next = (k0 + 32) < DM;

        // ---- stage next tile (A async, W via register transpose) ----
        v8bf bstage = {};
        if (has_next) {
            const u16* ap = A + (size_t)(blockM + arow) * DM + (k0 + 32) + aseg;
            u16* la = &sA[nxt][arow * SA_STRIDE + aseg];
#ifdef ASYNC_LDS
            async_cp16(ap, la);
            async_cp16(ap + 8, la + 8);
#else
            *(v8bf*)la       = ld8(ap);
            *(v8bf*)(la + 8) = ld8(ap + 8);
#endif
            bstage = ld8(W + (size_t)(k0 + 32 + bk) * DM + blockN + bn);
            if (k0 + 64 < DM)
                __builtin_prefetch(W + (size_t)(k0 + 64 + bk) * DM + blockN + bn, 0, 2);
        }

        // ---- compute current tile ----
        v16bf afr[2], bfr[2];
        #pragma unroll
        for (int i = 0; i < 2; ++i) {
            const u16* p = &sA[cur][(wm * 32 + i * 16 + ln) * SA_STRIDE + lh * 8];
            afr[i] = cat16(ld8(p), ld8(p + 16));   // K {0..7,16..23} / {8..15,24..31}
        }
        #pragma unroll
        for (int j = 0; j < 2; ++j) {
            const u16* p = &sB[cur][(wn * 32 + j * 16 + ln) * SA_STRIDE + lh * 8];
            bfr[j] = cat16(ld8(p), ld8(p + 16));   // K {0..7,16..23} / {8..15,24..31}
        }
        #pragma unroll
        for (int i = 0; i < 2; ++i)
            #pragma unroll
            for (int j = 0; j < 2; ++j)
                acc[i][j] = wmma_bf16(afr[i], bfr[j], acc[i][j]);

        if (has_next) {
            #pragma unroll
            for (int j = 0; j < 8; ++j)
                sB[nxt][(bn + j) * SA_STRIDE + bk] = __builtin_bit_cast(u16, bstage[j]);
        }
#ifdef ASYNC_LDS
        wait_async_lds();
#endif
        __syncthreads();
        cur = nxt;
    }

    // ---- epilogue (+bias), f32 or bf16 output ----
    #pragma unroll
    for (int i = 0; i < 2; ++i) {
        #pragma unroll
        for (int j = 0; j < 2; ++j) {
            const int col = blockN + wn * 32 + j * 16 + ln;
            const float bv = bias[col];
            #pragma unroll
            for (int r = 0; r < 8; ++r) {
                const int row = blockM + wm * 32 + i * 16 + r + lh * 8;
                const float v = acc[i][j][r] + bv;
                if (outF) outF[(size_t)row * DM + col] = v;
                else      outB[(size_t)row * DM + col] = f2bf(v);
            }
        }
    }
}

__global__ __launch_bounds__(256) void qkv_gemm(const u16* __restrict__ Xb,
                                                const u16* Wqb, const u16* Wkb, const u16* Wvb,
                                                const float* bq, const float* bk, const float* bv,
                                                u16* Qb, u16* Kb, u16* Vb) {
    const u16* W; const float* bias; u16* out;
    if (blockIdx.z == 0)      { W = Wqb; bias = bq; out = Qb; }
    else if (blockIdx.z == 1) { W = Wkb; bias = bk; out = Kb; }
    else                      { W = Wvb; bias = bv; out = Vb; }
    gemm_core(Xb, W, bias, nullptr, out);
}

__global__ __launch_bounds__(256) void out_proj(const u16* __restrict__ Ab,
                                                const u16* __restrict__ Wob,
                                                const float* __restrict__ bo,
                                                float* __restrict__ out) {
    gemm_core(Ab, Wob, bo, out, nullptr);
}

// ---------------------------------------------------------------------------
// Flash attention: grid (S/128, H, B), 256 threads (8 waves).
// Each wave owns 16 q-rows; Q frags held in registers for all 32 key blocks.
// K staged row-major (async when available), V staged transposed [depth][key];
// P round-trips through wave-private LDS (C-layout -> A-layout) for P@V.
// ---------------------------------------------------------------------------
#define KV_STRIDE 72
__global__ __launch_bounds__(256) void flash_attn(const u16* __restrict__ Qb,
                                                  const u16* __restrict__ Kb,
                                                  const u16* __restrict__ Vb,
                                                  u16* __restrict__ Ab) {
    __shared__ u16 sK [64 * KV_STRIDE];        // [key][depth]
    __shared__ u16 sVt[64 * KV_STRIDE];        // [depth][key]
    __shared__ u16 sP [8][16 * KV_STRIDE];     // per-wave P tile [qrow][key]

    const int tid  = threadIdx.x;
    const int lane = tid & 31;
    const int wave = tid >> 5;
    const int lh   = lane >> 4;
    const int ln   = lane & 15;
    const int h    = blockIdx.y;
    const int b    = blockIdx.z;
    const int qBase   = blockIdx.x * 128 + wave * 16;
    const int rowbase = b * SEQ;              // row offset in [B*S]
    const int cbase   = h * DEPTH;            // column offset in 1024

    // Q fragments (held for the whole loop)
    v16bf aq[2];
    {
        const u16* qp = Qb + (size_t)(rowbase + qBase + ln) * DM + cbase;
        #pragma unroll
        for (int c = 0; c < 2; ++c) {
            const u16* p = qp + c * 32 + lh * 8;
            aq[c] = cat16(ld8(p), ld8(p + 16));
        }
    }

    v8f o[4] = {};
    float mrow[8], lrow[8];
    #pragma unroll
    for (int r = 0; r < 8; ++r) { mrow[r] = -1e30f; lrow[r] = 0.f; }

    const int skey = tid >> 2;            // 0..63
    const int sseg = (tid & 3) * 16;      // depth segment

    for (int kb = 0; kb < SEQ; kb += 64) {
        // ---- stage K block (row-major), async when available ----
        const u16* kp = Kb + (size_t)(rowbase + kb + skey) * DM + cbase + sseg;
        u16* lk = sK + skey * KV_STRIDE + sseg;
#ifdef ASYNC_LDS
        async_cp16(kp, lk);
        async_cp16(kp + 8, lk + 8);
#else
        *(v8bf*)lk       = ld8(kp);
        *(v8bf*)(lk + 8) = ld8(kp + 8);
#endif
        // ---- stage V block transposed -> sVt[depth][key] ----
        const u16* vp = Vb + (size_t)(rowbase + kb + skey) * DM + cbase + sseg;
        v8bf v0 = ld8(vp), v1 = ld8(vp + 8);
        #pragma unroll
        for (int j = 0; j < 8; ++j) {
            sVt[(sseg + j)     * KV_STRIDE + skey] = __builtin_bit_cast(u16, v0[j]);
            sVt[(sseg + 8 + j) * KV_STRIDE + skey] = __builtin_bit_cast(u16, v1[j]);
        }
        if (kb + 64 < SEQ) {
            __builtin_prefetch(Kb + (size_t)(rowbase + kb + 64 + skey) * DM + cbase + sseg, 0, 2);
            __builtin_prefetch(Vb + (size_t)(rowbase + kb + 64 + skey) * DM + cbase + sseg, 0, 2);
        }
#ifdef ASYNC_LDS
        wait_async_lds();
#endif
        __syncthreads();

        // ---- S = Q * K^T, 16 q-rows x 64 keys (scaled by 1/sqrt(depth)) ----
        v8f sfr[4];
        #pragma unroll
        for (int j = 0; j < 4; ++j) {
            v8f acc = {};
            #pragma unroll
            for (int c = 0; c < 2; ++c) {
                const u16* bp = sK + (j * 16 + ln) * KV_STRIDE + c * 32 + lh * 8;
                v16bf bf = cat16(ld8(bp), ld8(bp + 16));
                acc = wmma_bf16(aq[c], bf, acc);
            }
            #pragma unroll
            for (int e = 0; e < 8; ++e) acc[e] = acc[e] * 0.125f;
            sfr[j] = acc;
        }

        // ---- online softmax (rows r / r+8 in lane halves; 16-lane shuffles) ----
        #pragma unroll
        for (int r = 0; r < 8; ++r) {
            float mx = fmaxf(fmaxf(sfr[0][r], sfr[1][r]), fmaxf(sfr[2][r], sfr[3][r]));
            #pragma unroll
            for (int off = 1; off < 16; off <<= 1)
                mx = fmaxf(mx, __shfl_xor(mx, off, 32));
            const float mnew  = fmaxf(mrow[r], mx);
            const float alpha = __expf(mrow[r] - mnew);
            float rsum = 0.f;
            #pragma unroll
            for (int j = 0; j < 4; ++j) {
                const float p = __expf(sfr[j][r] - mnew);
                sfr[j][r] = p;
                rsum += p;
            }
            #pragma unroll
            for (int off = 1; off < 16; off <<= 1)
                rsum += __shfl_xor(rsum, off, 32);
            lrow[r] = lrow[r] * alpha + rsum;
            mrow[r] = mnew;
            #pragma unroll
            for (int j = 0; j < 4; ++j) o[j][r] = o[j][r] * alpha;
        }

        // ---- P: C-layout -> LDS -> A-layout (wave-private; DS is in-order) ----
        u16* pw = &sP[wave][0];
        #pragma unroll
        for (int j = 0; j < 4; ++j)
            #pragma unroll
            for (int r = 0; r < 8; ++r)
                pw[(r + lh * 8) * KV_STRIDE + j * 16 + ln] = f2bf(sfr[j][r]);

        // ---- O += P * V ----
        #pragma unroll
        for (int j = 0; j < 4; ++j) {
            #pragma unroll
            for (int c = 0; c < 2; ++c) {
                const u16* ap = pw + ln * KV_STRIDE + c * 32 + lh * 8;
                v16bf af = cat16(ld8(ap), ld8(ap + 16));
                const u16* bp = sVt + (j * 16 + ln) * KV_STRIDE + c * 32 + lh * 8;
                v16bf bf = cat16(ld8(bp), ld8(bp + 16));
                o[j] = wmma_bf16(af, bf, o[j]);
            }
        }
        __syncthreads();
    }

    // ---- normalize and write attention output (bf16, [B*S][1024]) ----
    #pragma unroll
    for (int j = 0; j < 4; ++j) {
        const int col = cbase + j * 16 + ln;
        #pragma unroll
        for (int r = 0; r < 8; ++r) {
            const int row = rowbase + qBase + r + lh * 8;
            Ab[(size_t)row * DM + col] = f2bf(o[j][r] / lrow[r]);
        }
    }
}

// ---------------------------------------------------------------------------
// Launch
// ---------------------------------------------------------------------------
extern "C" void kernel_launch(void* const* d_in, const int* in_sizes, int n_in,
                              void* d_out, int out_size, void* d_ws, size_t ws_size,
                              hipStream_t stream) {
    const float* X  = (const float*)d_in[0];
    const float* Wq = (const float*)d_in[1];
    const float* bq = (const float*)d_in[2];
    const float* Wk = (const float*)d_in[3];
    const float* bk = (const float*)d_in[4];
    const float* Wv = (const float*)d_in[5];
    const float* bv = (const float*)d_in[6];
    const float* Wo = (const float*)d_in[7];
    const float* bo = (const float*)d_in[8];

    // workspace layout (u16 elements): 48 MiB total
    u16* Xb  = (u16*)d_ws;                   // 4096*1024
    u16* Wqb = Xb  + (size_t)MROWS * DM;
    u16* Wkb = Wqb + (size_t)DM * DM;
    u16* Wvb = Wkb + (size_t)DM * DM;
    u16* Wob = Wvb + (size_t)DM * DM;
    u16* Qb  = Wob + (size_t)DM * DM;
    u16* Kb  = Qb  + (size_t)MROWS * DM;
    u16* Vb  = Kb  + (size_t)MROWS * DM;
    u16* Ab  = Vb  + (size_t)MROWS * DM;

    const int nX = MROWS * DM;   // 4M
    const int nW = DM * DM;      // 1M
    cvt_bf16<<<nX / 4 / 256, 256, 0, stream>>>(X,  Xb,  nX);
    cvt_bf16<<<nW / 4 / 256, 256, 0, stream>>>(Wq, Wqb, nW);
    cvt_bf16<<<nW / 4 / 256, 256, 0, stream>>>(Wk, Wkb, nW);
    cvt_bf16<<<nW / 4 / 256, 256, 0, stream>>>(Wv, Wvb, nW);
    cvt_bf16<<<nW / 4 / 256, 256, 0, stream>>>(Wo, Wob, nW);

    // Q/K/V projections: grid (N/64, M/128, 3)
    qkv_gemm<<<dim3(DM / 64, MROWS / 128, 3), 256, 0, stream>>>(
        Xb, Wqb, Wkb, Wvb, bq, bk, bv, Qb, Kb, Vb);

    // flash attention: grid (S/128, H, B)
    flash_attn<<<dim3(SEQ / 128, HEADS, BATCH), 256, 0, stream>>>(Qb, Kb, Vb, Ab);

    // output projection -> f32 d_out
    out_proj<<<dim3(DM / 64, MROWS / 128, 1), 256, 0, stream>>>(
        Ab, Wob, bo, (float*)d_out);
}
